// MoERouter_6416681140831
// MI455X (gfx1250) — matrix operand order repaired and checked by reading, not compiled, not run on hardware
//
#include <hip/hip_runtime.h>
#include <hip/hip_bf16.h>

typedef __attribute__((ext_vector_type(16))) __bf16 v16bf;
typedef __attribute__((ext_vector_type(8)))  __bf16 v8bf;
typedef __attribute__((ext_vector_type(8)))  float  v8f;

#define NTOK        16384
#define HDIM        2048
#define NEXP        64
#define KC          64
#define NCHUNK      (HDIM / KC)
#define TOK_PER_WG  128
#define TOK_PER_WV  16
#define LDS_PAD     8
#define XROW        (KC + LDS_PAD)

// fp32 -> bf16 round-to-nearest-even, pure integer
static __device__ __forceinline__ unsigned int f32_to_bf16(float f) {
    unsigned int u = __float_as_uint(f);
    u += 0x7FFFu + ((u >> 16) & 1u);
    return u >> 16;
}

static __device__ __forceinline__ v16bf load_frag(const unsigned short* p0,
                                                  const unsigned short* p1) {
    const v8bf lo = *(const v8bf*)p0;
    const v8bf hi = *(const v8bf*)p1;
    return __builtin_shufflevector(lo, hi, 0,1,2,3,4,5,6,7,8,9,10,11,12,13,14,15);
}

__global__ void zero_acc_kernel(float* __restrict__ p) {
    int i = blockIdx.x * blockDim.x + threadIdx.x;
    if (i < 2 * NEXP + 1) p[i] = 0.0f;
}

__global__ void convert_w_kernel(const float* __restrict__ Wg,
                                 unsigned short* __restrict__ Wb) {
    int i = blockIdx.x * blockDim.x + threadIdx.x;
    if (i < NEXP * HDIM) Wb[i] = (unsigned short)f32_to_bf16(Wg[i]);
}

__launch_bounds__(256)
__global__ void moe_router_kernel(const float* __restrict__ X,
                                  const unsigned short* __restrict__ Wb,
                                  float* __restrict__ out,
                                  float* __restrict__ gcounts,
                                  float* __restrict__ gpsum,
                                  float* __restrict__ gz) {
    // Staging (double-buffered) aliased with the logit spill buffer: staging is
    // dead after the K loop, and the last loop barrier separates the lifetimes.
    __shared__ union SharedU {
        struct {
            alignas(16) unsigned short xs[2][TOK_PER_WG][XROW];
            alignas(16) unsigned short ws[2][NEXP][XROW];
        } st;
        float lg[8][TOK_PER_WV][NEXP + 1];
    } shm;
    __shared__ float hist[NEXP];
    __shared__ float psum[NEXP];
    __shared__ float zpart;

    const int tid  = threadIdx.x;
    const int lane = tid & 31;
    const int wave = tid >> 5;
    const int tok0 = blockIdx.x * TOK_PER_WG;
    const int wrow = wave * TOK_PER_WV;

    if (tid < NEXP) { hist[tid] = 0.0f; psum[tid] = 0.0f; }
    if (tid == 0)   zpart = 0.0f;

    const int m  = lane & 15;   // A row / B column within a 16-tile
    const int kh = lane >> 4;   // lane half -> K sub-range
    const int c4 = tid & 15;    // x staging: float4 column
    const int r16 = tid >> 4;   // x staging: base row (16 rows / pass)
    const int c8 = tid & 7;     // w staging: 16B chunk
    const int r32 = tid >> 3;   // w staging: base row (32 rows / pass)

    v8f acc[4] = {};
    float4 xr[8];

    // ---------------- prologue: stage chunk 0 into buffer 0 ----------------
    {
        #pragma unroll
        for (int rp = 0; rp < 2; ++rp) {
            const int row = rp * 32 + r32;
            const unsigned short* gp = Wb + (size_t)row * HDIM + c8 * 8;
            const unsigned int lds = (unsigned int)(size_t)&shm.st.ws[0][row][c8 * 8];
            asm volatile("global_load_async_to_lds_b128 %0, %1, off"
                         :: "v"(lds), "v"(gp) : "memory");
        }
        #pragma unroll
        for (int rp = 0; rp < 8; ++rp) {
            const int row = rp * 16 + r16;
            xr[rp] = *(const float4*)(X + (size_t)(tok0 + row) * HDIM + c4 * 4);
        }
        #pragma unroll
        for (int rp = 0; rp < 8; ++rp) {
            const int row = rp * 16 + r16;
            const unsigned int p0 = f32_to_bf16(xr[rp].x) | (f32_to_bf16(xr[rp].y) << 16);
            const unsigned int p1 = f32_to_bf16(xr[rp].z) | (f32_to_bf16(xr[rp].w) << 16);
            *(uint2*)&shm.st.xs[0][row][c4 * 4] = make_uint2(p0, p1);
        }
        asm volatile("s_wait_asynccnt 0" ::: "memory");
        __syncthreads();
    }

    // ---------------- pipelined K loop ----------------
    for (int i = 0; i < NCHUNK; ++i) {
        const int cur = i & 1;
        const int nxt = cur ^ 1;
        const int kcn = (i + 1) * KC;
        const bool has_next = (i + 1 < NCHUNK);   // uniform branch: EXEC stays full

        if (has_next) {
            // async DMA next W chunk straight into LDS (bf16, no conversion needed)
            #pragma unroll
            for (int rp = 0; rp < 2; ++rp) {
                const int row = rp * 32 + r32;
                const unsigned short* gp = Wb + (size_t)row * HDIM + kcn + c8 * 8;
                const unsigned int lds = (unsigned int)(size_t)&shm.st.ws[nxt][row][c8 * 8];
                asm volatile("global_load_async_to_lds_b128 %0, %1, off"
                             :: "v"(lds), "v"(gp) : "memory");
            }
            // next x chunk into registers; converted + stored after the WMMAs
            #pragma unroll
            for (int rp = 0; rp < 8; ++rp) {
                const int row = rp * 16 + r16;
                xr[rp] = *(const float4*)(X + (size_t)(tok0 + row) * HDIM + kcn + c4 * 4);
            }
        }

        // ---- fetch all fragments into independent registers, then 8 WMMAs
        v16bf A[2], B[2][4];
        #pragma unroll
        for (int kb2 = 0; kb2 < 2; ++kb2) {
            const int kb = kb2 * 32;
            A[kb2] = load_frag(&shm.st.xs[cur][wrow + m][kb + kh * 8],
                               &shm.st.xs[cur][wrow + m][kb + 16 + kh * 8]);
            #pragma unroll
            for (int t = 0; t < 4; ++t) {
                const int e = t * 16 + m;
                B[kb2][t] = load_frag(&shm.st.ws[cur][e][kb + kh * 8],
                                      &shm.st.ws[cur][e][kb + 16 + kh * 8]);
            }
        }
        #pragma unroll
        for (int kb2 = 0; kb2 < 2; ++kb2)
            #pragma unroll
            for (int t = 0; t < 4; ++t)
                acc[t] = __builtin_amdgcn_wmma_f32_16x16x32_bf16(
                    false, A[kb2], false, B[kb2][t], (short)0, acc[t], false, false);

        if (has_next) {
            #pragma unroll
            for (int rp = 0; rp < 8; ++rp) {
                const int row = rp * 16 + r16;
                const unsigned int p0 = f32_to_bf16(xr[rp].x) | (f32_to_bf16(xr[rp].y) << 16);
                const unsigned int p1 = f32_to_bf16(xr[rp].z) | (f32_to_bf16(xr[rp].w) << 16);
                *(uint2*)&shm.st.xs[nxt][row][c4 * 4] = make_uint2(p0, p1);
            }
        }
        asm volatile("s_wait_asynccnt 0" ::: "memory");
        __syncthreads();
    }

    // ---- spill logits (aliases staging; final loop barrier makes this safe)
    #pragma unroll
    for (int t = 0; t < 4; ++t)
        #pragma unroll
        for (int r = 0; r < 8; ++r)
            shm.lg[wave][kh * 8 + r][t * 16 + m] = acc[t][r];
    __syncthreads();

    // ---- softmax / top-2 / aux losses: lanes 0..15 each own one token
    if (lane < 16) {
        const int n = tok0 + wrow + lane;
        const float* L = &shm.lg[wave][lane][0];
        float mx = -3.4e38f;
        for (int e = 0; e < NEXP; ++e) mx = fmaxf(mx, L[e]);
        float s = 0.0f, b1 = -3.4e38f, b2 = -3.4e38f;
        int i1 = 0, i2 = 0;
        for (int e = 0; e < NEXP; ++e) {
            const float l = L[e];
            s += __expf(l - mx);
            if (l > b1)      { b2 = b1; i2 = i1; b1 = l; i1 = e; }
            else if (l > b2) { b2 = l; i2 = e; }
        }
        const float inv_s = 1.0f / s;
        for (int e = 0; e < NEXP; ++e)
            atomicAdd(&psum[e], __expf(L[e] - mx) * inv_s);
        const float lse = mx + __logf(s);
        atomicAdd(&zpart, lse * lse);
        atomicAdd(&hist[i1], 1.0f);
        const float p1 = __expf(b1 - mx), p2 = __expf(b2 - mx);
        const float dn = 1.0f / (p1 + p2);
        out[2 * n]                = p1 * dn;
        out[2 * n + 1]            = p2 * dn;
        out[2 * NTOK + 2 * n]     = (float)i1;
        out[2 * NTOK + 2 * n + 1] = (float)i2;
    }
    __syncthreads();
    if (tid < NEXP) {
        atomicAdd(&gcounts[tid], hist[tid]);
        atomicAdd(&gpsum[tid],  psum[tid]);
    }
    if (tid == 0) atomicAdd(gz, zpart);
}

__global__ void finalize_kernel(const float* __restrict__ gcounts,
                                const float* __restrict__ gpsum,
                                const float* __restrict__ gz,
                                float* __restrict__ out) {
    __shared__ float red[NEXP];
    const int e = threadIdx.x;
    const float c = gcounts[e];
    out[4 * NTOK + 2 + e] = c * (1.0f / NTOK);        // utilization
    red[e] = c * gpsum[e];
    __syncthreads();
    for (int s = 32; s > 0; s >>= 1) {
        if (e < s) red[e] += red[e + s];
        __syncthreads();
    }
    if (e == 0) {
        out[4 * NTOK]     = (float)NEXP * red[0] / ((float)NTOK * (float)NTOK); // LB loss
        out[4 * NTOK + 1] = gz[0] * (1.0f / NTOK);                              // z loss
    }
}

extern "C" void kernel_launch(void* const* d_in, const int* in_sizes, int n_in,
                              void* d_out, int out_size, void* d_ws, size_t ws_size,
                              hipStream_t stream) {
    const float* X  = (const float*)d_in[0];   // [4,4096,2048] fp32
    const float* Wg = (const float*)d_in[1];   // [64,2048] fp32
    float* out = (float*)d_out;

    unsigned short* Wb = (unsigned short*)d_ws;                       // 64*2048 bf16
    float* accbuf = (float*)((char*)d_ws + (size_t)NEXP * HDIM * 2);
    float* gcounts = accbuf;
    float* gpsum   = accbuf + NEXP;
    float* gz      = accbuf + 2 * NEXP;

    zero_acc_kernel<<<1, 256, 0, stream>>>(accbuf);
    convert_w_kernel<<<(NEXP * HDIM + 255) / 256, 256, 0, stream>>>(Wg, Wb);
    moe_router_kernel<<<NTOK / TOK_PER_WG, 256, 0, stream>>>(X, Wb, out, gcounts, gpsum, gz);
    finalize_kernel<<<1, NEXP, 0, stream>>>(gcounts, gpsum, gz, out);
}
